// GCN_37383395344580
// MI455X (gfx1250) — matrix operand order repaired and checked by reading, not compiled, not run on hardware
//
#include <hip/hip_runtime.h>

// ---------------------------------------------------------------------------
// GCN (3x GCNConv + global mean pool) for gfx1250 / MI455X.
// GEMMs use v_wmma_f32_16x16x32_bf16 (wave32 WMMA). Edge aggregation uses
// hardware f32 global atomics (L2-resident: feature matrices are 25.6MB,
// global L2 is 192MB).
// ---------------------------------------------------------------------------

typedef __attribute__((ext_vector_type(16))) __bf16 v16bf;
typedef __attribute__((ext_vector_type(8)))  __bf16 v8bf;
typedef __attribute__((ext_vector_type(8)))  float  v8f;

union AV16 {
    v16bf v;
    v8bf  h[2];
};

__device__ __forceinline__ void atomAddF32(float* p, float v) {
    // HW global_atomic_add_f32 (no CAS loop)
    unsafeAtomicAdd(p, v);
}

// ---------------- small utility kernels ----------------

__global__ void k_fill_f32(float* __restrict__ p, float v, int n) {
    int t = blockIdx.x * blockDim.x + threadIdx.x;
    if (t < n) p[t] = v;
}

__global__ void k_deg_edges(float* __restrict__ deg, const int* __restrict__ dst, int E) {
    int t = blockIdx.x * blockDim.x + threadIdx.x;
    if (t < E) atomAddF32(&deg[dst[t]], 1.0f);
}

__global__ void k_dinv(float* __restrict__ deg, int n) {
    int t = blockIdx.x * blockDim.x + threadIdx.x;
    if (t < n) {
        float d = deg[t];
        deg[t] = (d > 0.0f) ? rsqrtf(d) : 0.0f;
    }
}

__global__ void k_norm(float* __restrict__ nrm, const float* __restrict__ dinv,
                       const int* __restrict__ src, const int* __restrict__ dst, int E) {
    int t = blockIdx.x * blockDim.x + threadIdx.x;
    if (t < E) nrm[t] = dinv[src[t]] * dinv[dst[t]];
}

__global__ void k_f32_to_bf16(__bf16* __restrict__ dstp, const float* __restrict__ srcp,
                              int n, int relu) {
    int t = blockIdx.x * blockDim.x + threadIdx.x;
    if (t < n) {
        float v = srcp[t];
        if (relu) v = fmaxf(v, 0.0f);
        dstp[t] = (__bf16)v;
    }
}

// ---------------- WMMA GEMM: H[N,FOUT] = Xb[N,64] @ Wb[64,FOUT] ----------------
// One wave per 16x16 output tile; waves within a block cover the FOUT/16 n-tiles.
// A layout (16-bit A 16x32, contiguous K like B): lane half 0 holds K {kb+0..15},
// half 1 holds K {kb+16..31} for row M = lane%16.
// B layout: v16bf element i <-> K = kb + half*16 + i, N = lane%16.
// W is staged TRANSPOSED in LDS (sWT[col*64 + k]) so each lane's 16 B elements
// are one contiguous 32-byte run -> ds_load_b128 x2, no bf16 packing movs.

template <int FOUT>
__global__ void __launch_bounds__((FOUT / 16) * 32)
k_gemm_wmma(const __bf16* __restrict__ X, const float* __restrict__ W,
            float* __restrict__ H, int nrows) {
    __shared__ __align__(32) __bf16 sWT[FOUT * 64];
    const int tid = threadIdx.x;
    for (int i = tid; i < 64 * FOUT; i += blockDim.x) {
        int k = i / FOUT, c = i - k * FOUT;
        sWT[c * 64 + k] = (__bf16)W[i];
    }
    __syncthreads();

    const int lane = tid & 31;
    const int wave = tid >> 5;            // n-tile index
    const int half = lane >> 4;
    const int mloc = lane & 15;
    const int ncol = wave * 16 + mloc;

    int mrow = blockIdx.x * 16 + mloc;
    if (mrow > nrows - 1) mrow = nrows - 1;   // clamp loads; stores guarded below
    const __bf16* xrow = X + (size_t)mrow * 64;
    const __bf16* wcol = sWT + ncol * 64;

    v8f c = {};
#pragma unroll
    for (int kb = 0; kb < 64; kb += 32) {
        AV16 a;
        a.h[0] = *(const v8bf*)(xrow + kb + half * 16);       // K = kb+half*16 .. +7
        a.h[1] = *(const v8bf*)(xrow + kb + half * 16 + 8);   // K = kb+half*16+8 .. +15
        v16bf b = *(const v16bf*)(wcol + kb + half * 16);     // K = kb+half*16 .. +15
        c = __builtin_amdgcn_wmma_f32_16x16x32_bf16(
            /*neg_a=*/false, a.v, /*neg_b=*/false, b,
            /*c_mod=*/(short)0, c, /*reuse_a=*/false, /*reuse_b=*/false);
    }

    const int row0 = blockIdx.x * 16;
    if (row0 + 16 <= nrows) {
        // fast path: branch-free, coalesced b32 stores
        float* hp = H + (size_t)(row0 + 8 * half) * FOUT + ncol;
#pragma unroll
        for (int v = 0; v < 8; ++v) hp[(size_t)v * FOUT] = c[v];   // M = v + 8*half
    } else {
#pragma unroll
        for (int v = 0; v < 8; ++v) {
            int row = row0 + v + 8 * half;
            if (row < nrows) H[(size_t)row * FOUT + ncol] = c[v];
        }
    }
}

// ---------------- aggregation ----------------
// out[i,f] = b[f] + h[i,f]*dinv[i]^2   (self-loop term folded in)
__global__ void k_agg_init(float* __restrict__ out, const float* __restrict__ h,
                           const float* __restrict__ bias, const float* __restrict__ dinv,
                           int n, int F) {
    int t = blockIdx.x * blockDim.x + threadIdx.x;
    if (t >= n * F) return;
    int i = t / F, f = t - i * F;
    float di = dinv[i];
    out[t] = bias[f] + h[t] * di * di;
}

// out[dst[e], f] += h[src[e], f] * norm[e]   (2 features per thread)
__global__ void k_agg_edges(float* __restrict__ out, const float* __restrict__ h,
                            const float* __restrict__ nrm,
                            const int* __restrict__ src, const int* __restrict__ dst,
                            int E, int F) {
    int t = blockIdx.x * blockDim.x + threadIdx.x;
    const int per = F >> 1;                  // threads per edge (2 floats each)
    if (t >= E * per) return;
    int e = t / per;
    int f = (t - e * per) * 2;
    float w = nrm[e];
    int s = src[e], d = dst[e];
    const float2 hv = *(const float2*)(h + (size_t)s * F + f);
    float* op = out + (size_t)d * F + f;
    atomAddF32(op + 0, hv.x * w);
    atomAddF32(op + 1, hv.y * w);
}

// ---------------- mean pool ----------------
__global__ void k_pool_sum(float* __restrict__ pooled, const float* __restrict__ h,
                           const int* __restrict__ batch, int n) {
    int t = blockIdx.x * blockDim.x + threadIdx.x;
    if (t >= n * 32) return;
    int i = t >> 5, f = t & 31;
    atomAddF32(&pooled[batch[i] * 32 + f], h[t]);
}

__global__ void k_pool_cnt(float* __restrict__ cnt, const int* __restrict__ batch, int n) {
    int t = blockIdx.x * blockDim.x + threadIdx.x;
    if (t < n) atomAddF32(&cnt[batch[t]], 1.0f);
}

__global__ void k_pool_div(float* __restrict__ outp, const float* __restrict__ pooled,
                           const float* __restrict__ cnt, int n) {
    int t = blockIdx.x * blockDim.x + threadIdx.x;
    if (t < n) outp[t] = pooled[t] / fmaxf(cnt[t >> 5], 1.0f);
}

// ---------------------------------------------------------------------------

extern "C" void kernel_launch(void* const* d_in, const int* in_sizes, int n_in,
                              void* d_out, int out_size, void* d_ws, size_t ws_size,
                              hipStream_t stream) {
    const float* x     = (const float*)d_in[0];
    const int*   ei    = (const int*)d_in[1];
    const int*   batch = (const int*)d_in[2];
    const float* W1 = (const float*)d_in[4];
    const float* b1 = (const float*)d_in[5];
    const float* W2 = (const float*)d_in[6];
    const float* b2 = (const float*)d_in[7];
    const float* W3 = (const float*)d_in[8];
    const float* b3 = (const float*)d_in[9];
    float* outp = (float*)d_out;

    const int N = in_sizes[0] / 64;        // 100000
    const int E = in_sizes[1] / 2;         // 1600000
    const int G = out_size / 32;           // 64
    const int* srcp = ei;
    const int* dstp = ei + E;

    // workspace carve-out (256B aligned)
    char* ws = (char*)d_ws;
    size_t off = 0;
    auto carve = [&](size_t bytes) {
        void* p = ws + off;
        off += (bytes + 255) & ~(size_t)255;
        return p;
    };
    float*  dinv   = (float*)carve((size_t)N * sizeof(float));
    float*  nrm    = (float*)carve((size_t)E * sizeof(float));
    __bf16* xb     = (__bf16*)carve((size_t)N * 64 * sizeof(__bf16));
    float*  h      = (float*)carve((size_t)N * 64 * sizeof(float));
    float*  agg    = (float*)carve((size_t)N * 64 * sizeof(float));
    float*  pooled = (float*)carve((size_t)G * 32 * sizeof(float));
    float*  cnt    = (float*)carve((size_t)G * sizeof(float));
    (void)ws_size; (void)n_in;

    const int T = 256;
    auto blks = [](long long n, int t) { return (unsigned)((n + t - 1) / t); };
    const int mtiles = (N + 15) / 16;

    // ---- normalization: deg = 1 (self loop) + in-degree; dinv = rsqrt; norm ----
    k_fill_f32<<<blks(N, T), T, 0, stream>>>(dinv, 1.0f, N);
    k_deg_edges<<<blks(E, T), T, 0, stream>>>(dinv, dstp, E);
    k_dinv<<<blks(N, T), T, 0, stream>>>(dinv, N);
    k_norm<<<blks(E, T), T, 0, stream>>>(nrm, dinv, srcp, dstp, E);

    // ---- layer 1 ----
    k_f32_to_bf16<<<blks((long long)N * 64, T), T, 0, stream>>>(xb, x, N * 64, 0);
    k_gemm_wmma<64><<<mtiles, 128, 0, stream>>>(xb, W1, h, N);
    k_agg_init<<<blks((long long)N * 64, T), T, 0, stream>>>(agg, h, b1, dinv, N, 64);
    k_agg_edges<<<blks((long long)E * 32, T), T, 0, stream>>>(agg, h, nrm, srcp, dstp, E, 64);
    k_f32_to_bf16<<<blks((long long)N * 64, T), T, 0, stream>>>(xb, agg, N * 64, 1); // ReLU

    // ---- layer 2 ----
    k_gemm_wmma<64><<<mtiles, 128, 0, stream>>>(xb, W2, h, N);
    k_agg_init<<<blks((long long)N * 64, T), T, 0, stream>>>(agg, h, b2, dinv, N, 64);
    k_agg_edges<<<blks((long long)E * 32, T), T, 0, stream>>>(agg, h, nrm, srcp, dstp, E, 64);
    k_f32_to_bf16<<<blks((long long)N * 64, T), T, 0, stream>>>(xb, agg, N * 64, 0);

    // ---- layer 3 (FOUT = 32) ----
    k_gemm_wmma<32><<<mtiles, 64, 0, stream>>>(xb, W3, h, N);
    k_agg_init<<<blks((long long)N * 32, T), T, 0, stream>>>(agg, h, b3, dinv, N, 32);
    k_agg_edges<<<blks((long long)E * 16, T), T, 0, stream>>>(agg, h, nrm, srcp, dstp, E, 32);

    // ---- global mean pool ----
    k_fill_f32<<<blks(G * 32, T), T, 0, stream>>>(pooled, 0.0f, G * 32);
    k_fill_f32<<<1, T, 0, stream>>>(cnt, 0.0f, G);
    k_pool_sum<<<blks((long long)N * 32, T), T, 0, stream>>>(pooled, agg, batch, N);
    k_pool_cnt<<<blks(N, T), T, 0, stream>>>(cnt, batch, N);
    k_pool_div<<<blks(G * 32, T), T, 0, stream>>>(outp, pooled, cnt, G * 32);
}
